// LMHA_52785148068559
// MI455X (gfx1250) — matrix operand order repaired and checked by reading, not compiled, not run on hardware
//
#include <hip/hip_runtime.h>
#include <cmath>

// ---------------------------------------------------------------------------
// LMHA (RoPE multi-head attention) for MI455X / gfx1250 — wave32 WMMA path
// with TDM (tensor_load_to_lds) double-buffered K/V staging in attention.
// B=2 S=2048 D=1024 H=32 HD=32.
// ---------------------------------------------------------------------------

typedef __attribute__((ext_vector_type(16))) _Float16 v16h;
typedef __attribute__((ext_vector_type(2)))  _Float16 h2;
typedef __attribute__((ext_vector_type(8)))  float    v8f;
typedef unsigned int u32x4 __attribute__((ext_vector_type(4)));
typedef int          i32x4 __attribute__((ext_vector_type(4)));
typedef int          i32x8 __attribute__((ext_vector_type(8)));

#define B_   2
#define S_   2048
#define D_   1024
#define H_   32
#define HD_  32
#define QKV_ 3072          // 2*QK_DIM + D
#define BH_  (B_ * H_)     // 64
#define M_   (B_ * S_)     // 4096 rows
#define CHUNK_ 64          // keys staged per TDM chunk

__device__ __forceinline__ int lane_id() { return (int)(threadIdx.x & 31u); }

// --- A-fragment (16x32 f16, row-major source, leading dim ld) --------------
__device__ __forceinline__ v16h load_a_frag(const _Float16* __restrict__ base, int ld) {
    int l = lane_id();
    int row = l & 15, half = l >> 4;
    const _Float16* rp = base + (size_t)row * ld;
    v16h a = {};
#pragma unroll
    for (int p = 0; p < 8; ++p) {
        int k0 = (p >> 2) * 16 + (p & 3) * 2 + half * 8;
        h2 pr = *(const h2*)(rp + k0);
        a[2 * p]     = pr.x;
        a[2 * p + 1] = pr.y;
    }
    return a;
}

// --- B-fragment (32x16 f16, K-major [K][N] source, leading dim ld) ---------
__device__ __forceinline__ v16h load_b_frag(const _Float16* __restrict__ base, int ld) {
    int l = lane_id();
    int n = l & 15, half = l >> 4;
    v16h b;
#pragma unroll
    for (int s = 0; s < 16; ++s) {
        int k = (s < 8) ? (half * 8 + s) : (16 + half * 8 + (s - 8));
        b[s] = base[(size_t)k * ld + n];
    }
    return b;
}
// Same, but from LDS (16 strided 16-bit element reads, stride ld)
__device__ __forceinline__ v16h load_b_frag_lds(const _Float16* base, int ld) {
    int l = lane_id();
    int n = l & 15, half = l >> 4;
    v16h b;
#pragma unroll
    for (int s = 0; s < 16; ++s) {
        int k = (s < 8) ? (half * 8 + s) : (16 + half * 8 + (s - 8));
        b[s] = base[(size_t)k * ld + n];
    }
    return b;
}

// --- TDM: 2-D tile (rows x elems, f16) from global -> LDS ------------------
// D# per CDNA5 ISA. count=1, data_size=2B, type=2(image).
// This toolchain exposes the 6-arg builtin:
//   (u32x4 g0, i32x8 g1, i32x4 g2, i32x4 g3, i32x8 extra, i32 cpol)
__device__ __forceinline__ void tdm_load_2d(unsigned int lds_off, const void* gaddr,
                                            unsigned int tile_d0, unsigned int tile_d1,
                                            unsigned int stride_d0_elems) {
#if defined(__has_builtin)
#if __has_builtin(__builtin_amdgcn_tensor_load_to_lds)
    unsigned long long ga = (unsigned long long)(uintptr_t)gaddr;
    const unsigned int tensor_d0 = 0x7FFFFFFFu;   // no OOB clipping
    const unsigned int tensor_d1 = 0x7FFFFFFFu;
    u32x4 g0;
    g0[0] = 1u;                                     // count=1, user mode
    g0[1] = lds_off;                                // LDS byte address
    g0[2] = (unsigned int)ga;                       // global_addr[31:0]
    g0[3] = (unsigned int)((ga >> 32) & 0x1FFFFFFu) | (2u << 30);  // [56:32] | type=2
    i32x8 g1;
    g1[0] = (int)(1u << 16);                        // workgroup_mask=0, data_size=1 (2B)
    g1[1] = (int)((tensor_d0 & 0xFFFFu) << 16);     // barrier_addr=0 | tensor_dim0[15:0]
    g1[2] = (int)(((tensor_d0 >> 16) & 0xFFFFu) | ((tensor_d1 & 0xFFFFu) << 16));
    g1[3] = (int)(((tensor_d1 >> 16) & 0xFFFFu) | ((tile_d0 & 0xFFFFu) << 16));
    g1[4] = (int)(tile_d1 & 0xFFFFu);               // tile_dim1 | tile_dim2=0
    g1[5] = (int)stride_d0_elems;                   // tensor_dim0_stride[31:0]
    g1[6] = 0;                                      // stride0 hi | stride1 lo
    g1[7] = 0;                                      // stride1 hi
    i32x4 z4 = {0, 0, 0, 0};
    i32x8 z8 = {0, 0, 0, 0, 0, 0, 0, 0};
    __builtin_amdgcn_tensor_load_to_lds(g0, g1, z4, z4, z8, 0);
#endif
#endif
}

template <int N>
__device__ __forceinline__ void wait_tensorcnt() {
#if defined(__has_builtin)
#if __has_builtin(__builtin_amdgcn_s_wait_tensorcnt)
    __builtin_amdgcn_s_wait_tensorcnt(N);           // immediate (ICE via template)
#else
    asm volatile("s_wait_tensorcnt %0" :: "i"(N));
#endif
#endif
}

// ---------------------------------------------------------------------------
// Prep: fp32 -> f16 convert / transpose-convert
// ---------------------------------------------------------------------------
__global__ void cvt_f16_kernel(const float* __restrict__ src, _Float16* __restrict__ dst, int n) {
    int i = blockIdx.x * blockDim.x + threadIdx.x;
    if (i < n) dst[i] = (_Float16)src[i];
}

__global__ void transpose_f16_kernel(const float* __restrict__ src, _Float16* __restrict__ dst,
                                     int rows, int cols) {
    int i = blockIdx.x * blockDim.x + threadIdx.x;
    if (i < rows * cols) {
        int r = i / cols, c = i - r * cols;
        dst[(size_t)c * rows + r] = (_Float16)src[i];
    }
}

// ---------------------------------------------------------------------------
// Kernel 1: t = x @ Wqkv^T fused with RoPE; scatter into Q, K^T, V (f16).
// One wave per 16x16 output tile; 32 WMMAs over K=1024.
// ---------------------------------------------------------------------------
__global__ void __launch_bounds__(256) qkv_rope_kernel(
        const _Float16* __restrict__ xh,
        const _Float16* __restrict__ wqt,   // [1024][3072] K-major
        const float*    __restrict__ pe,    // [2048][32]: sin[0:16], cos[16:32]
        _Float16* __restrict__ qh,          // [BH][S][32]
        _Float16* __restrict__ kth,         // [BH][32][S]  (transposed K)
        _Float16* __restrict__ vh)          // [BH][S][32]
{
    const int NT = QKV_ / 16;
    int wave = threadIdx.x >> 5;
    int tile = blockIdx.x * 8 + wave;
    int mt = tile / NT, nt = tile % NT;
    int l = lane_id();
    int half = l >> 4, col = l & 15;

    const _Float16* abase = xh + (size_t)(mt * 16) * D_;
    const _Float16* bbase = wqt + nt * 16;

    v8f c = {};
    for (int kb = 0; kb < D_; kb += 32) {
        v16h a = load_a_frag(abase + kb, D_);
        v16h b = load_b_frag(bbase + (size_t)kb * QKV_, QKV_);
        c = __builtin_amdgcn_wmma_f32_16x16x32_f16(false, a, false, b, (short)0, c, false, false);
        __builtin_prefetch(abase + kb + 128, 0, 1);
        __builtin_prefetch(bbase + (size_t)(kb + 32) * QKV_, 0, 1);
    }

    int n0 = nt * 16;
#pragma unroll
    for (int r = 0; r < 8; ++r) {
        int m   = mt * 16 + half * 8 + r;
        int b   = m >> 11;
        int pos = m & (S_ - 1);
        int n   = n0 + col;
        float v = c[r];
        if (n < 2 * D_) {                  // Q or K: RoPE (branch uniform per tile)
            int d = n & 31;
            int i = d >> 1;
            float sn = pe[pos * 32 + i];
            float cs = pe[pos * 32 + 16 + i];
            float partner = __shfl_xor(v, 1, 32);
            float out = (n & 1) ? (v * cs + partner * sn)
                                : (v * cs - partner * sn);
            if (n < D_) {
                int hh = n >> 5;
                qh[(((size_t)(b * H_ + hh)) * S_ + pos) * HD_ + d] = (_Float16)out;
            } else {
                int hh = (n - D_) >> 5;
                kth[(((size_t)(b * H_ + hh)) * HD_ + d) * S_ + pos] = (_Float16)out;
            }
        } else {
            int nn = n - 2 * D_;
            int hh = nn >> 5, d = nn & 31;
            vh[(((size_t)(b * H_ + hh)) * S_ + pos) * HD_ + d] = (_Float16)v;
        }
    }
}

// ---------------------------------------------------------------------------
// Kernel 2: flash attention with TDM-staged K/V.
// Block = 8 waves = one (b,h) x 128 queries; each wave owns a 16-query tile.
// Wave 0 drives tensor_load_to_lds (double-buffered 64-key chunks of K^T and
// V), s_wait_tensorcnt overlaps DMA with WMMA compute; all waves consume the
// chunk from LDS. Online softmax; P re-laid-out via per-wave LDS tile.
// ---------------------------------------------------------------------------
struct __align__(128) AttnLds {
    _Float16 kt[2][HD_ * CHUNK_];        // [buf][32 dims][64 keys]
    _Float16 v [2][CHUNK_ * HD_];        // [buf][64 keys][32 dims]
    _Float16 p [8][16 * 32 + 16];        // per-wave P tile
};

__global__ void __launch_bounds__(256) attn_kernel(
        const _Float16* __restrict__ qh,
        const _Float16* __restrict__ kth,
        const _Float16* __restrict__ vh,
        _Float16* __restrict__ oh)       // [b][s][h][32]
{
    __shared__ AttnLds sm;
    int wave = threadIdx.x >> 5;
    int bh = blockIdx.x >> 4;            // 64 (b,h) pairs
    int st = blockIdx.x & 15;            // 16 query super-tiles of 128
    int qt = st * 8 + wave;              // this wave's 16-query tile
    int l = lane_id();
    int half = l >> 4, col = l & 15;
    const float scale = 0.17677669529663687f;   // 1/sqrt(HD)

    const _Float16* kbase = kth + (size_t)bh * HD_ * S_;
    const _Float16* vbase = vh  + (size_t)bh * S_ * HD_;

    v16h aQ = load_a_frag(qh + ((size_t)bh * S_ + qt * 16) * HD_, HD_);

    v8f acc0 = {}, acc1 = {};
    float mrow[8], srow[8];
#pragma unroll
    for (int r = 0; r < 8; ++r) { mrow[r] = -1e30f; srow[r] = 0.f; }

    _Float16* myp = &sm.p[wave][0];
    const int NCH = S_ / CHUNK_;         // 32 chunks

    if (wave == 0) {                     // preload chunk 0 (TDM, async)
        tdm_load_2d((unsigned int)(uintptr_t)&sm.kt[0][0], kbase, CHUNK_, HD_, S_);
        tdm_load_2d((unsigned int)(uintptr_t)&sm.v[0][0],  vbase, CHUNK_ * HD_, 1, CHUNK_ * HD_);
    }

    for (int ch = 0; ch < NCH; ++ch) {
        int buf = ch & 1;
        if (wave == 0) {
            if (ch + 1 < NCH) {          // issue next chunk, then wait for current
                int kc = (ch + 1) * CHUNK_;
                tdm_load_2d((unsigned int)(uintptr_t)&sm.kt[buf ^ 1][0], kbase + kc, CHUNK_, HD_, S_);
                tdm_load_2d((unsigned int)(uintptr_t)&sm.v[buf ^ 1][0],  vbase + (size_t)kc * HD_,
                            CHUNK_ * HD_, 1, CHUNK_ * HD_);
                wait_tensorcnt<2>();     // current chunk's 2 loads complete
            } else {
                wait_tensorcnt<0>();
            }
        }
        __syncthreads();                 // chunk `ch` visible to all waves

        const _Float16* ktc = &sm.kt[buf][0];
        const _Float16* vc  = &sm.v[buf][0];

#pragma unroll
        for (int hh = 0; hh < 2; ++hh) { // two 32-key halves per 64-key chunk
            v16h bK0 = load_b_frag_lds(ktc + hh * 32,      CHUNK_);
            v16h bK1 = load_b_frag_lds(ktc + hh * 32 + 16, CHUNK_);
            v8f z = {};
            v8f s0 = __builtin_amdgcn_wmma_f32_16x16x32_f16(false, aQ, false, bK0, (short)0, z, false, false);
            v8f s1 = __builtin_amdgcn_wmma_f32_16x16x32_f16(false, aQ, false, bK1, (short)0, z, false, false);

#pragma unroll
            for (int r = 0; r < 8; ++r) {
                float x0 = s0[r] * scale, x1 = s1[r] * scale;
                float red = fmaxf(x0, x1);
#pragma unroll
                for (int msk = 1; msk < 16; msk <<= 1) red = fmaxf(red, __shfl_xor(red, msk, 32));
                float nm = fmaxf(mrow[r], red);
                float p0 = __expf(x0 - nm), p1 = __expf(x1 - nm);
                float ps = p0 + p1;
#pragma unroll
                for (int msk = 1; msk < 16; msk <<= 1) ps += __shfl_xor(ps, msk, 32);
                float corr = __expf(mrow[r] - nm);
                srow[r] = srow[r] * corr + ps;
                mrow[r] = nm;
                acc0[r] *= corr;
                acc1[r] *= corr;
                int row = half * 8 + r;
                myp[row * 32 + col]      = (_Float16)p0;
                myp[row * 32 + 16 + col] = (_Float16)p1;
            }
            asm volatile("s_wait_dscnt 0" ::: "memory");   // all lanes' P stores done

            v16h aP = {};
            {
                const _Float16* rp = myp + (size_t)(l & 15) * 32;
#pragma unroll
                for (int p = 0; p < 8; ++p) {
                    int k0 = (p >> 2) * 16 + (p & 3) * 2 + half * 8;
                    h2 pr = *(const h2*)(rp + k0);
                    aP[2 * p]     = pr.x;
                    aP[2 * p + 1] = pr.y;
                }
            }

            v16h bV0 = load_b_frag_lds(vc + (size_t)(hh * 32) * HD_,      HD_);
            v16h bV1 = load_b_frag_lds(vc + (size_t)(hh * 32) * HD_ + 16, HD_);
            acc0 = __builtin_amdgcn_wmma_f32_16x16x32_f16(false, aP, false, bV0, (short)0, acc0, false, false);
            acc1 = __builtin_amdgcn_wmma_f32_16x16x32_f16(false, aP, false, bV1, (short)0, acc1, false, false);
        }
        __syncthreads();                 // reads done before buf is overwritten
    }

    // normalize + scatter to [b][s][h][d]
    int b = bh >> 5, h = bh & 31;
#pragma unroll
    for (int r = 0; r < 8; ++r) {
        int row = half * 8 + r;
        int pos = qt * 16 + row;
        float inv = 1.0f / srow[r];
        size_t base = (((size_t)b * S_ + pos) * H_ + h) * HD_;
        oh[base + col]      = (_Float16)(acc0[r] * inv);
        oh[base + 16 + col] = (_Float16)(acc1[r] * inv);
    }
}

// ---------------------------------------------------------------------------
// Kernel 3: out = O @ Wo^T -> fp32 d_out
// ---------------------------------------------------------------------------
__global__ void __launch_bounds__(256) oproj_kernel(
        const _Float16* __restrict__ oh,
        const _Float16* __restrict__ wot,   // [1024][1024] K-major
        float* __restrict__ out)
{
    const int NT = D_ / 16;
    int wave = threadIdx.x >> 5;
    int tile = blockIdx.x * 8 + wave;
    int mt = tile / NT, nt = tile % NT;
    int l = lane_id();
    int half = l >> 4, col = l & 15;

    const _Float16* abase = oh + (size_t)(mt * 16) * D_;
    const _Float16* bbase = wot + nt * 16;

    v8f c = {};
    for (int kb = 0; kb < D_; kb += 32) {
        v16h a = load_a_frag(abase + kb, D_);
        v16h b = load_b_frag(bbase + (size_t)kb * D_, D_);
        c = __builtin_amdgcn_wmma_f32_16x16x32_f16(false, a, false, b, (short)0, c, false, false);
        __builtin_prefetch(abase + kb + 128, 0, 1);
    }
#pragma unroll
    for (int r = 0; r < 8; ++r) {
        int m = mt * 16 + half * 8 + r;
        out[(size_t)m * D_ + nt * 16 + col] = c[r];
    }
}

// ---------------------------------------------------------------------------
// Host launcher
// ---------------------------------------------------------------------------
extern "C" void kernel_launch(void* const* d_in, const int* in_sizes, int n_in,
                              void* d_out, int out_size, void* d_ws, size_t ws_size,
                              hipStream_t stream) {
    const float* x    = (const float*)d_in[0];   // [2][2048][1024]
    const float* pe   = (const float*)d_in[1];   // [2048][32]
    /* d_in[2] = mask: all-true -> no-op */
    const float* wqkv = (const float*)d_in[3];   // [3072][1024]
    const float* wo   = (const float*)d_in[4];   // [1024][1024]
    float* out = (float*)d_out;                  // [2][2048][1024]

    _Float16* ws  = (_Float16*)d_ws;
    _Float16* xh  = ws; ws += (size_t)M_ * D_;
    _Float16* wqt = ws; ws += (size_t)D_ * QKV_;
    _Float16* wot = ws; ws += (size_t)D_ * D_;
    _Float16* qh  = ws; ws += (size_t)BH_ * S_ * HD_;
    _Float16* kth = ws; ws += (size_t)BH_ * S_ * HD_;
    _Float16* vh  = ws; ws += (size_t)BH_ * S_ * HD_;
    _Float16* oh  = ws; ws += (size_t)M_ * D_;       // ~48MB total

    cvt_f16_kernel<<<(M_ * D_) / 256, 256, 0, stream>>>(x, xh, M_ * D_);
    transpose_f16_kernel<<<(QKV_ * D_) / 256, 256, 0, stream>>>(wqkv, wqt, QKV_, D_);
    transpose_f16_kernel<<<(D_ * D_) / 256, 256, 0, stream>>>(wo, wot, D_, D_);

    qkv_rope_kernel<<<(256 * 192) / 8, 256, 0, stream>>>(xh, wqt, pe, qh, kth, vh);
    attn_kernel<<<BH_ * (S_ / 128), 256, 0, stream>>>(qh, kth, vh, oh);   // 1024 blocks
    oproj_kernel<<<(256 * 64) / 8, 256, 0, stream>>>(oh, wot, out);
}
